// MultiHeadAttentionBlock_87153476370798
// MI455X (gfx1250) — matrix operand order repaired and checked by reading, not compiled, not run on hardware
//
#include <hip/hip_runtime.h>
#include <hip/hip_bf16.h>
#include <math.h>

// ---------------------------------------------------------------------------
// MultiHeadAttentionBlock for MI455X (gfx1250, wave32, WMMA + TDM)
// B=4, T=1024, D=1024, H=16, hd=64, NUM_BUCKETS=320, MAX_DIST=800
// ---------------------------------------------------------------------------

typedef _Float16 f16;
typedef __attribute__((ext_vector_type(16))) _Float16 v16h;
typedef __attribute__((ext_vector_type(8)))  _Float16 v8h;
typedef __attribute__((ext_vector_type(8)))  float    v8f;
typedef __attribute__((ext_vector_type(4)))  unsigned int v4u;
typedef __attribute__((ext_vector_type(8)))  int v8i;
typedef __attribute__((ext_vector_type(4)))  int v4i;

#define TT    1024
#define DD    1024
#define BB    4
#define HH    16
#define HDIM  64
#define MROWS (BB*TT)   // 4096

#if defined(__has_builtin)
# if __has_builtin(__builtin_amdgcn_tensor_load_to_lds) && \
     __has_builtin(__builtin_amdgcn_s_wait_tensorcnt)
#  define USE_TDM 1
# endif
#endif
#ifndef USE_TDM
# define USE_TDM 0
#endif

// ---- WMMA helpers ---------------------------------------------------------
// 16-bit A-matrix 16x32 layout: lane (m = lane&15, half = lane>>4)
// holds K = half*16 + [0..15] contiguous (matches B fragments).
__device__ __forceinline__ v16h mk_a_frag(const f16* rowk, int half) {
  v8h lo = *(const v8h*)(rowk + half * 16);
  v8h hi = *(const v8h*)(rowk + half * 16 + 8);
  return __builtin_shufflevector(lo, hi, 0,1,2,3,4,5,6,7,8,9,10,11,12,13,14,15);
}
// 16-bit B-matrix 32x16: lane n holds column n, K = half*16 + [0..15] contiguous.
__device__ __forceinline__ v16h ld_b_frag(const f16* col_row_k0, int half) {
  return *(const v16h*)(col_row_k0 + half * 16);
}
__device__ __forceinline__ v8f wmma16(v16h a, v16h b, v8f c) {
  return __builtin_amdgcn_wmma_f32_16x16x32_f16(
      /*neg_a=*/false, a, /*neg_b=*/false, b,
      /*c_mod=*/(short)0, c, /*reuse_a=*/false, /*reuse_b=*/false);
}

// ---- Tensor Data Mover: 2D tile (tile_d1 rows x tile_d0 elems) -> LDS -----
// D# per ISA 8.3/8.4: count=1, type=2 (image), data_size=1 (2-byte elems),
// tensor_dim0 = row capacity, tensor_dim0_stride = row pitch (elements).
#if USE_TDM
__device__ __forceinline__ void tdm_load_2d_f16(unsigned lds_off, const void* gptr,
                                                unsigned tile_d0, unsigned tile_d1,
                                                unsigned stride0_elems) {
  const unsigned long long ga = (unsigned long long)(size_t)gptr;
  const unsigned td0 = stride0_elems;   // dim0 length >= tile_d0
  const unsigned td1 = 0x100000u;       // generous dim1 length (OOB never hit)
  v4u g0;
  g0[0] = 1u;                                               // count=1, user mode
  g0[1] = lds_off;                                          // lds_addr (bytes)
  g0[2] = (unsigned)ga;                                     // global_addr lo
  g0[3] = ((unsigned)(ga >> 32) & 0x01FFFFFFu) | (2u << 30);// addr hi | type=2
  v8i g1;
  g1[0] = (int)(1u << 16);                                  // mask=0, data_size=2B
  g1[1] = (int)((td0 & 0xFFFFu) << 16);                     // abar=0 | dim0[15:0]
  g1[2] = (int)((td0 >> 16) | ((td1 & 0xFFFFu) << 16));     // dim0[31:16] | dim1 lo
  g1[3] = (int)((td1 >> 16) | (tile_d0 << 16));             // dim1 hi | tile_dim0
  g1[4] = (int)tile_d1;                                     // tile_dim1 | tile_dim2=0
  g1[5] = (int)stride0_elems;                               // dim0_stride lo32
  g1[6] = 0;                                                // dim0_stride hi | dim1_stride lo
  g1[7] = 0;                                                // dim1_stride hi
  const v4i z4 = {0, 0, 0, 0};
#if __clang_major__ >= 23
  const v8i z8 = {0, 0, 0, 0, 0, 0, 0, 0};
  __builtin_amdgcn_tensor_load_to_lds(g0, g1, z4, z4, z8, 0);
#else
  __builtin_amdgcn_tensor_load_to_lds(g0, g1, z4, z4, 0);
#endif
}
#endif

// ---- f32 -> f16 weight conversion ----------------------------------------
__global__ void __launch_bounds__(256)
cvt_kernel(const float* __restrict__ src, f16* __restrict__ dst, int n) {
  int i = blockIdx.x * 256 + threadIdx.x;
  if (i < n) dst[i] = (f16)src[i];
}

// ---- LayerNorm: one block (256 thr) per row, f16 output -------------------
__global__ void __launch_bounds__(256)
ln_kernel(const float* __restrict__ x, const float* __restrict__ lw,
          const float* __restrict__ lb, f16* __restrict__ h16) {
  const int row = blockIdx.x;
  const int tid = threadIdx.x;
  const float* xr = x + (size_t)row * DD;
  float4 v = *(const float4*)(xr + tid * 4);
  float s = v.x + v.y + v.z + v.w;
  float q = v.x*v.x + v.y*v.y + v.z*v.z + v.w*v.w;
#pragma unroll
  for (int m = 16; m; m >>= 1) { s += __shfl_xor(s, m, 32); q += __shfl_xor(q, m, 32); }
  __shared__ float red[16];
  if ((tid & 31) == 0) { red[tid >> 5] = s; red[8 + (tid >> 5)] = q; }
  __syncthreads();
  float ts = 0.f, tq = 0.f;
#pragma unroll
  for (int i = 0; i < 8; ++i) { ts += red[i]; tq += red[8 + i]; }
  const float mu   = ts * (1.f / DD);
  const float var  = tq * (1.f / DD) - mu * mu;
  const float rstd = rsqrtf(var + 1e-5f);
  const int d = tid * 4;
  f16* hr = h16 + (size_t)row * DD + d;
  hr[0] = (f16)(((v.x - mu) * rstd) * lw[d+0] + lb[d+0]);
  hr[1] = (f16)(((v.y - mu) * rstd) * lw[d+1] + lb[d+1]);
  hr[2] = (f16)(((v.z - mu) * rstd) * lw[d+2] + lb[d+2]);
  hr[3] = (f16)(((v.w - mu) * rstd) * lw[d+3] + lb[d+3]);
}

// ---- WMMA GEMM: C(4096x1024) = A(f16) @ W(f16)^T (+bias, mode epilogue) ---
// mode 0: Q  -> out16[m][n] = (acc+b)*scale        (t-major)
// mode 1: K  -> out16[m][n] =  acc+b               (t-major)
// mode 2: V  -> out16 transposed per head: [(b*H+h)*64+d][t]
// mode 3: O  -> out32[m][n] = acc+b+resid[m][n]    (final projection)
__global__ void __launch_bounds__(256)
gemm_kernel(const f16* __restrict__ A, const f16* __restrict__ W,
            const float* __restrict__ bias, int mode, float scale,
            f16* __restrict__ out16, float* __restrict__ out32,
            const float* __restrict__ resid) {
  const int wave = blockIdx.x * 8 + (threadIdx.x >> 5);
  const int lane = threadIdx.x & 31;
  const int r16  = lane & 15, half = lane >> 4;
  const int m0 = (wave >> 4) * 16;   // 256 M-tiles
  const int n0 = (wave & 15) * 64;   // 16 N-supertiles of 64

  v8f acc[4] = {};
  const f16* arow = A + (size_t)(m0 + r16) * DD;
  const f16* wr0  = W + (size_t)(n0 + r16) * DD;
  for (int k = 0; k < DD; k += 32) {
    if (k + 64 < DD) __builtin_prefetch(arow + k + 64, 0, 1);  // global_prefetch_b8
    const v16h a  = mk_a_frag(arow + k, half);
    const v16h b0 = ld_b_frag(wr0 + (size_t)0 * 16 * DD + k, half);
    const v16h b1 = ld_b_frag(wr0 + (size_t)1 * 16 * DD + k, half);
    const v16h b2 = ld_b_frag(wr0 + (size_t)2 * 16 * DD + k, half);
    const v16h b3 = ld_b_frag(wr0 + (size_t)3 * 16 * DD + k, half);
    acc[0] = wmma16(a, b0, acc[0]);
    acc[1] = wmma16(a, b1, acc[1]);
    acc[2] = wmma16(a, b2, acc[2]);
    acc[3] = wmma16(a, b3, acc[3]);
  }
#pragma unroll
  for (int j = 0; j < 4; ++j) {
    const int n  = n0 + j * 16 + r16;
    const float bv = bias[n];
#pragma unroll
    for (int r = 0; r < 8; ++r) {
      const int m = m0 + r + half * 8;            // C layout: M = r + half*8
      float val = acc[j][r] + bv;
      if (mode == 0) {
        out16[(size_t)m * DD + n] = (f16)(val * scale);
      } else if (mode == 1) {
        out16[(size_t)m * DD + n] = (f16)val;
      } else if (mode == 2) {
        const int b_ = m >> 10, t = m & 1023;
        const int hh = n >> 6,  d = n & 63;
        out16[((size_t)((b_ * HH + hh) * HDIM + d)) * TT + t] = (f16)val;
      } else {
        out32[(size_t)m * DD + n] = val + resid[(size_t)m * DD + n];
      }
    }
  }
}

// ---- relative bucket table (T x T) ---------------------------------------
__global__ void __launch_bounds__(256)
bucket_kernel(int* __restrict__ bucket) {
  const int idx = blockIdx.x * 256 + threadIdx.x;
  const int t = idx >> 10, sc = idx & 1023;
  const int rel = sc - t;
  int bk = (rel > 0) ? 160 : 0;           // half = NUM_BUCKETS/2 = 160
  const int n = rel < 0 ? -rel : rel;
  if (n < 80) {                           // max_exact = 80
    bk += n;
  } else {
    const float nf = (float)n;
    int large = 80 + (int)(__logf(nf * (1.0f / 80.0f)) * (80.0f / __logf(10.0f)));
    bk += (large < 159 ? large : 159);
  }
  bucket[idx] = bk;
}

// ---- pos_bias_out: (B*H, T, T) written straight into d_out ----------------
__global__ void __launch_bounds__(256)
posbias_kernel(const int* __restrict__ bucket, const float* __restrict__ rel_emb,
               float* __restrict__ pbout) {
  const size_t idx = (size_t)blockIdx.x * 256 + threadIdx.x;
  const int sc = (int)(idx & 1023);
  const size_t rr = idx >> 10;
  const int t  = (int)(rr & 1023);
  const int bh = (int)(rr >> 10);
  const int h  = bh & 15;
  pbout[idx] = rel_emb[bucket[t * TT + sc] * HH + h];
}

// ---- GREP gate: scalar per (b,h,t) ----------------------------------------
__global__ void __launch_bounds__(256)
gate_kernel(const f16* __restrict__ q16, const float* __restrict__ gw,
            const float* __restrict__ gb, const float* __restrict__ ga,
            float* __restrict__ gate) {
  const int idx = blockIdx.x * 256 + threadIdx.x;   // (b*H+h)*T + t
  const int t  = idx & 1023;
  const int bh = idx >> 10;
  const int h  = bh & 15, b = bh >> 4;
  const f16* qrow = q16 + ((size_t)(b * TT + t)) * DD + h * HDIM;
  float acc[8];
#pragma unroll
  for (int j = 0; j < 8; ++j) acc[j] = gb[j];
  for (int d = 0; d < HDIM; ++d) {
    // q16 stores q*scaling; ql = q*alpha/scaling = raw*32 -> *256 here
    const float ql = (float)qrow[d] * 256.0f;
#pragma unroll
    for (int j = 0; j < 8; ++j) acc[j] += ql * gw[j * HDIM + d];
  }
  const float s0 = acc[0] + acc[1] + acc[2] + acc[3];
  const float s1 = acc[4] + acc[5] + acc[6] + acc[7];
  const float g0 = 1.f / (1.f + __expf(-s0));
  const float g1 = 1.f / (1.f + __expf(-s1));
  gate[idx] = g0 * (g1 * ga[h] - 1.0f) + 2.0f;
}

// ---- flash attention: WMMA q.kT and p.v, streaming softmax ----------------
// grid (T/64, B*H), block 128 (4 waves); each wave owns 16 query rows.
// All 4 waves share the same K/V tiles, so wave 0 stages them into LDS with
// the Tensor Data Mover (tensor_load_to_lds + s_wait_tensorcnt), and all
// waves build WMMA B-fragments from LDS.
__global__ void __launch_bounds__(128)
attn_kernel(const f16* __restrict__ q16, const f16* __restrict__ k16,
            const f16* __restrict__ vT, const float* __restrict__ gate,
            const float* __restrict__ pb, f16* __restrict__ o16) {
  __shared__ __align__(64) f16 pbuf[4][16][32];   // C-layout -> A-layout repack
#if USE_TDM
  __shared__ __align__(128) f16 kst[32][64];      // K tile: [s_local][kd]
  __shared__ __align__(128) f16 vst[64][32];      // V tile: [d][s_local]
#endif
  const int w    = threadIdx.x >> 5;
  const int lane = threadIdx.x & 31;
  const int r16  = lane & 15, half = lane >> 4;
  const int bh = blockIdx.y;
  const int b  = bh >> 4, h = bh & 15;
  const int qt0 = (blockIdx.x * 4 + w) * 16;

  const f16*   qbase = q16 + ((size_t)b * TT) * DD + h * HDIM;
  const f16*   kbase = k16 + ((size_t)b * TT) * DD + h * HDIM;
  const f16*   vbase = vT + ((size_t)bh * HDIM) * TT;
  const float* pbb   = pb + (size_t)bh * TT * TT;
  const float* grow  = gate + (size_t)bh * TT;

  const f16* qrow = qbase + (size_t)(qt0 + r16) * DD;
  const v16h a0 = mk_a_frag(qrow, half);        // K = 0..31 of head dim
  const v16h a1 = mk_a_frag(qrow + 32, half);   // K = 32..63

  float m_r[8], l_r[8], g_r[8];
  v8f o[4] = {};
#pragma unroll
  for (int r = 0; r < 8; ++r) {
    m_r[r] = -1e30f; l_r[r] = 0.f;
    g_r[r] = grow[qt0 + r + half * 8];
  }

#if USE_TDM
  const unsigned koff = (unsigned)(size_t)(void*)&kst[0][0];
  const unsigned voff = (unsigned)(size_t)(void*)&vst[0][0];
#endif

  for (int s0 = 0; s0 < TT; s0 += 32) {
#if USE_TDM
    __syncthreads();                      // previous tile fully consumed
    if (w == 0) {
      // K tile: 32 rows (s) x 64 elems (kd), row pitch D elements
      tdm_load_2d_f16(koff, kbase + (size_t)s0 * DD, HDIM, 32, DD);
      // V tile: 64 rows (d) x 32 elems (t),  row pitch T elements
      tdm_load_2d_f16(voff, vbase + s0, 32, HDIM, TT);
      __builtin_amdgcn_s_wait_tensorcnt((short)0);
    }
    __syncthreads();                      // publish LDS tiles to all waves
    const f16* kr0 = &kst[r16][0];
    const f16* kr1 = &kst[16 + r16][0];
#else
    const f16* kr0 = kbase + (size_t)(s0 + r16) * DD;
    const f16* kr1 = kbase + (size_t)(s0 + 16 + r16) * DD;
#endif
    // ---- scores for two 16-wide s tiles, K-chained over head dim ----
    v8f c0 = {}, c1 = {};
    c0 = wmma16(a0, ld_b_frag(kr0, half), c0);
    c0 = wmma16(a1, ld_b_frag(kr0 + 32, half), c0);
    c1 = wmma16(a0, ld_b_frag(kr1, half), c1);
    c1 = wmma16(a1, ld_b_frag(kr1 + 32, half), c1);

    // ---- + gate*pos_bias, online softmax (row reduce over 16 lanes) ----
#pragma unroll
    for (int r = 0; r < 8; ++r) {
      const float* pbr = pbb + (size_t)(qt0 + r + half * 8) * TT + s0 + r16;
      float s0v = c0[r] + g_r[r] * pbr[0];
      float s1v = c1[r] + g_r[r] * pbr[16];
      float mx = fmaxf(s0v, s1v);
      mx = fmaxf(mx, __shfl_xor(mx, 1, 32));
      mx = fmaxf(mx, __shfl_xor(mx, 2, 32));
      mx = fmaxf(mx, __shfl_xor(mx, 4, 32));
      mx = fmaxf(mx, __shfl_xor(mx, 8, 32));
      const float mnew = fmaxf(m_r[r], mx);
      const float corr = __expf(m_r[r] - mnew);
      const float p0 = __expf(s0v - mnew);
      const float p1 = __expf(s1v - mnew);
      float rs = p0 + p1;
      rs += __shfl_xor(rs, 1, 32);
      rs += __shfl_xor(rs, 2, 32);
      rs += __shfl_xor(rs, 4, 32);
      rs += __shfl_xor(rs, 8, 32);
      l_r[r] = l_r[r] * corr + rs;
      m_r[r] = mnew;
#pragma unroll
      for (int dt = 0; dt < 4; ++dt) o[dt][r] *= corr;
      pbuf[w][r + half * 8][r16]      = (f16)p0;
      pbuf[w][r + half * 8][16 + r16] = (f16)p1;
    }

    // ---- p @ v ----
    const v16h pa = mk_a_frag(&pbuf[w][r16][0], half);
#pragma unroll
    for (int dt = 0; dt < 4; ++dt) {
#if USE_TDM
      const f16* vrow = &vst[dt * 16 + r16][0];
#else
      const f16* vrow = vbase + (size_t)(dt * 16 + r16) * TT + s0;
#endif
      o[dt] = wmma16(pa, *(const v16h*)(vrow + half * 16), o[dt]);
    }
  }

  // ---- normalize and write f16 output (t-major, per head slot) ----
  f16* orow = o16 + ((size_t)(b * TT)) * DD + h * HDIM;
#pragma unroll
  for (int dt = 0; dt < 4; ++dt)
#pragma unroll
    for (int r = 0; r < 8; ++r) {
      const float val = o[dt][r] / l_r[r];
      orow[(size_t)(qt0 + r + half * 8) * DD + dt * 16 + r16] = (f16)val;
    }
}

// ---------------------------------------------------------------------------
extern "C" void kernel_launch(void* const* d_in, const int* in_sizes, int n_in,
                              void* d_out, int out_size, void* d_ws, size_t ws_size,
                              hipStream_t stream) {
  (void)in_sizes; (void)n_in; (void)out_size; (void)ws_size;
  const float* x       = (const float*)d_in[0];
  const float* ln_w    = (const float*)d_in[1];
  const float* ln_b    = (const float*)d_in[2];
  const float* wq      = (const float*)d_in[3];
  const float* bq      = (const float*)d_in[4];
  const float* wk      = (const float*)d_in[5];
  const float* bk      = (const float*)d_in[6];
  const float* wv      = (const float*)d_in[7];
  const float* bv      = (const float*)d_in[8];
  const float* wo      = (const float*)d_in[9];
  const float* bo      = (const float*)d_in[10];
  const float* rel_emb = (const float*)d_in[11];
  const float* grep_w  = (const float*)d_in[12];
  const float* grep_b  = (const float*)d_in[13];
  const float* grep_a  = (const float*)d_in[14];

  float* out   = (float*)d_out;                       // (B,T,D)
  float* pbout = out + (size_t)BB * TT * DD;          // (B*H,T,T)

  char* ws = (char*)d_ws;                             // ~53 MB used
  f16*  h16  = (f16*)(ws);
  f16*  q16  = (f16*)(ws + ( 8ull << 20));
  f16*  k16  = (f16*)(ws + (16ull << 20));
  f16*  vT   = (f16*)(ws + (24ull << 20));
  f16*  o16  = (f16*)(ws + (32ull << 20));
  f16*  wq16 = (f16*)(ws + (40ull << 20));
  f16*  wk16 = (f16*)(ws + (42ull << 20));
  f16*  wv16 = (f16*)(ws + (44ull << 20));
  f16*  wo16 = (f16*)(ws + (46ull << 20));
  float* gate   = (float*)(ws + (48ull << 20));
  int*   bucket = (int*)(ws + (49ull << 20));

  const int NW = DD * DD;
  cvt_kernel<<<NW / 256, 256, 0, stream>>>(wq, wq16, NW);
  cvt_kernel<<<NW / 256, 256, 0, stream>>>(wk, wk16, NW);
  cvt_kernel<<<NW / 256, 256, 0, stream>>>(wv, wv16, NW);
  cvt_kernel<<<NW / 256, 256, 0, stream>>>(wo, wo16, NW);

  ln_kernel<<<MROWS, 256, 0, stream>>>(x, ln_w, ln_b, h16);

  // Q scaled by hd^-0.5 = 0.125; V written transposed per head.
  gemm_kernel<<<512, 256, 0, stream>>>(h16, wq16, bq, 0, 0.125f, q16, nullptr, nullptr);
  gemm_kernel<<<512, 256, 0, stream>>>(h16, wk16, bk, 1, 1.0f,  k16, nullptr, nullptr);
  gemm_kernel<<<512, 256, 0, stream>>>(h16, wv16, bv, 2, 1.0f,  vT,  nullptr, nullptr);

  gate_kernel<<<(BB * HH * TT) / 256, 256, 0, stream>>>(q16, grep_w, grep_b, grep_a, gate);
  bucket_kernel<<<(TT * TT) / 256, 256, 0, stream>>>(bucket);
  posbias_kernel<<<(unsigned)(((size_t)BB * HH * TT * TT) / 256), 256, 0, stream>>>(
      bucket, rel_emb, pbout);

  attn_kernel<<<dim3(TT / 64, BB * HH), 128, 0, stream>>>(q16, k16, vT, gate, pbout, o16);

  gemm_kernel<<<512, 256, 0, stream>>>(o16, wo16, bo, 3, 1.0f, nullptr, out, x);
}
